// cheb_conv_with_Att_static_687194768239
// MI455X (gfx1250) — matrix operand order repaired and checked by reading, not compiled, not run on hardware
//
#include <hip/hip_runtime.h>

typedef __bf16 v16bf __attribute__((ext_vector_type(16)));
typedef float  v8f   __attribute__((ext_vector_type(8)));
typedef unsigned int u32t;
typedef u32t v4u __attribute__((ext_vector_type(4)));
typedef int  v8i __attribute__((ext_vector_type(8)));
typedef int  v4i __attribute__((ext_vector_type(4)));

#define V_DIM 2048
#define T_DIM 12
#define F_DIM 16
#define B_DIM 8
#define K_CHEB 3
#define C_DIM 64
#define M_TOT 192   // T*F

#if __has_builtin(__builtin_amdgcn_tensor_load_to_lds) && __has_builtin(__builtin_amdgcn_s_wait_tensorcnt)
#define USE_TDM 1
#else
#define USE_TDM 0
#endif

// ---------------------------------------------------------------------------
// Stage 0: XT[b][m][u] = bf16(x[b][t][u][f]),  m = t*16 + f
// ---------------------------------------------------------------------------
__global__ void build_xt(const float* __restrict__ x, __bf16* __restrict__ xt) {
  size_t i = (size_t)blockIdx.x * blockDim.x + threadIdx.x;
  const size_t N = (size_t)B_DIM * M_TOT * V_DIM;
  if (i >= N) return;
  int u = (int)(i & (V_DIM - 1));
  size_t r = i >> 11;
  int m = (int)(r % M_TOT);
  int b = (int)(r / M_TOT);
  int t = m >> 4, f = m & 15;
  xt[i] = (__bf16)x[(((size_t)b * T_DIM + t) * V_DIM + u) * F_DIM + f];
}

#if USE_TDM
// Issue a 2D TDM load: tile_dim1 x tile_dim0 bf16 elements, row stride `stride0`
// elements, from `gptr` into LDS offset `lds_off`. D# packing per ISA 8.3/8.4.
// amdgpu-toolchain (clang-23) 6-arg form: (v4u, v8i, v4i, v4i, v8i, i32 cpol).
__device__ __forceinline__ void tdm_load_2d(u32t lds_off, const void* gptr,
                                            u32t tile0, u32t tile1,
                                            u32t tens0, u32t tens1,
                                            u32t stride0) {
  unsigned long long ga = (unsigned long long)gptr;
  v4u g0 = { 1u,                                   // count=1, user descriptor
             lds_off,                              // lds_addr (bytes)
             (u32t)(ga & 0xffffffffu),             // global_addr[31:0]
             (u32t)((ga >> 32) & 0x1ffffffu) | (2u << 30) }; // [56:32] | type=2
  v8i g1 = { (int)(1u << 16),                      // data_size=1 (2 bytes)
             (int)((tens0 & 0xffffu) << 16),       // tensor_dim0[15:0]
             (int)((tens0 >> 16) | ((tens1 & 0xffffu) << 16)),
             (int)((tens1 >> 16) | (tile0 << 16)), // tile_dim0
             (int)(tile1 & 0xffffu),               // tile_dim1 (tile_dim2=0)
             (int)stride0,                         // tensor_dim0_stride[31:0]
             0, 0 };                               // stride hi, dim1_stride=0
  v4i gz4 = { 0, 0, 0, 0 };
  v8i gz8 = { 0, 0, 0, 0, 0, 0, 0, 0 };
  __builtin_amdgcn_tensor_load_to_lds(g0, g1, gz4, gz4, gz8, 0);
}
#endif

// ---------------------------------------------------------------------------
// Fused kernel: per (b, m_half, v_block of 32):
//   D_k[m=tf, v] = sum_u XT[m,u] * (cheb[k,u,v]*Att[b,u,v])   (bf16 WMMA, f32 acc)
//   out[b,t,v,c] = relu( sum_{k,f} D_k[t*16+f, v] * theta[k,f,c] )  (bf16 WMMA)
// ---------------------------------------------------------------------------
__launch_bounds__(128)
__global__ void cheb_fused(const __bf16* __restrict__ xt,
                           const float*  __restrict__ Att,
                           const float*  __restrict__ cheb,
                           const float*  __restrict__ theta,
                           float*        __restrict__ out) {
  // double-buffered masked-S tiles, transposed [v][u], padded row stride 40
  __shared__ __align__(16) __bf16 S[2][K_CHEB][32][40];
#if USE_TDM
  // double-buffered A tile: 96 rows (this m-half) x 32 u, filled by the TDM
  __shared__ __align__(16) __bf16 AX[2][96][32];
#endif
  // theta transposed: TT[c][kf], kf padded 48 -> 64 with zeros
  __shared__ __align__(16) __bf16 TT[64][64];
  // per-wave stage-2 A staging: Rt[v(16)][kf(64)], padded stride 72
  __shared__ __align__(16) __bf16 RT[4][16][72];

  const int tid  = threadIdx.x;
  const int w    = tid >> 5;
  const int lane = tid & 31;
  const int hi   = lane >> 4;
  const int l16  = lane & 15;
  const int vb   = blockIdx.x * 32;
  const int mh   = blockIdx.y;
  const int b    = blockIdx.z;

  const int n_tile = w & 1;
  const int m_set  = w >> 1;

  for (int idx = tid; idx < 64 * 64; idx += 128) {
    int kf = idx & 63, c = idx >> 6;
    float tv = (kf < 48) ? theta[kf * 64 + c] : 0.f;
    TT[c][kf] = (__bf16)tv;
  }

  v8f acc[3][K_CHEB] = {};

  const __bf16* xt_b  = xt  + (size_t)b * M_TOT * V_DIM + (size_t)(mh * 96) * V_DIM;
  const float*  att_b = Att + (size_t)b * V_DIM * V_DIM;

  // ---- staging helpers (inlined twice below) ----
#define STAGE_S(sb, uu0)                                                        \
  do {                                                                          \
    _Pragma("unroll")                                                           \
    for (int rep = 0; rep < 2; ++rep) {                                         \
      int q  = rep * 128 + tid;                                                 \
      int uu = q >> 3;                                                          \
      int vq = (q & 7) * 4;                                                     \
      const float4 a4 = *(const float4*)(att_b + (size_t)((uu0) + uu) * V_DIM + vb + vq); \
      _Pragma("unroll")                                                         \
      for (int k = 0; k < K_CHEB; ++k) {                                        \
        const float4 c4 = *(const float4*)(cheb + ((size_t)k * V_DIM + ((uu0) + uu)) * V_DIM + vb + vq); \
        S[sb][k][vq + 0][uu] = (__bf16)(a4.x * c4.x);                           \
        S[sb][k][vq + 1][uu] = (__bf16)(a4.y * c4.y);                           \
        S[sb][k][vq + 2][uu] = (__bf16)(a4.z * c4.z);                           \
        S[sb][k][vq + 3][uu] = (__bf16)(a4.w * c4.w);                           \
      }                                                                         \
    }                                                                           \
  } while (0)

#if USE_TDM
#define STAGE_A(ab, uu0)                                                        \
  do {                                                                          \
    if (w == 0) {                                                               \
      u32t lo = (u32t)(unsigned long long)(&AX[ab][0][0]);                      \
      tdm_load_2d(lo, xt_b + (uu0), 32u, 96u, 2048u, 96u, (u32t)V_DIM);         \
    }                                                                           \
  } while (0)
#define WAIT_A()                                                                \
  do { if (w == 0) __builtin_amdgcn_s_wait_tensorcnt(0); } while (0)
#else
#define STAGE_A(ab, uu0) do { } while (0)
#define WAIT_A()         do { } while (0)
#endif

  // prologue: fill buffer 0
  STAGE_A(0, 0);
  STAGE_S(0, 0);
  WAIT_A();
  __syncthreads();

  for (int u0 = 0; u0 < V_DIM; u0 += 32) {
    const int p = (u0 >> 5) & 1;
    if (u0 + 32 < V_DIM) {
      STAGE_A(1 - p, u0 + 32);
      STAGE_S(1 - p, u0 + 32);
    }
    if (u0 + 32 < V_DIM) {
      __builtin_prefetch(att_b + (size_t)(u0 + 32 + (tid >> 2)) * V_DIM + vb, 0, 1);
      __builtin_prefetch(cheb  + (size_t)((tid % 3) * V_DIM + u0 + 32 + (tid >> 2)) * V_DIM + vb, 0, 1);
    }

    // ---- B fragments from S[p] ----
    v16bf bf[K_CHEB];
    #pragma unroll
    for (int k = 0; k < K_CHEB; ++k) {
      const uint4* pp = (const uint4*)&S[p][k][n_tile * 16 + l16][hi * 16];
      union { uint4 q[2]; v16bf v; } un;
      un.q[0] = pp[0]; un.q[1] = pp[1];
      bf[k] = un.v;
    }
    // ---- A fragments + WMMA ----
    #pragma unroll
    for (int i = 0; i < 3; ++i) {
      union { uint4 q[2]; v16bf v; } ua;
#if USE_TDM
      const int ml = m_set * 48 + i * 16 + l16;        // row within the 96-row tile
      const uint4* pa = (const uint4*)&AX[p][ml][hi * 16];
      ua.q[0] = pa[0];   // K = off .. off+7
      ua.q[1] = pa[1];   // K = off+8 .. off+15
#else
      const int mrow = m_set * 48 + i * 16 + l16;
      const uint4* pa = (const uint4*)(xt_b + (size_t)mrow * V_DIM + u0 + hi * 16);
      ua.q[0] = pa[0];
      ua.q[1] = pa[1];
#endif
      v16bf af = ua.v;
      #pragma unroll
      for (int k = 0; k < K_CHEB; ++k) {
        acc[i][k] = __builtin_amdgcn_wmma_f32_16x16x32_bf16(
            false, af, false, bf[k], (short)0, acc[i][k], false, false);
      }
    }
    WAIT_A();          // next A tile landed (issuing wave only)
    __syncthreads();   // publish next buffers / protect current ones
  }

  // ---- preload stage-2 theta B fragments (K=kf x N=c) ----
  v16bf tb[4][2];
  #pragma unroll
  for (int ct = 0; ct < 4; ++ct) {
    #pragma unroll
    for (int ks = 0; ks < 2; ++ks) {
      const uint4* pp = (const uint4*)&TT[ct * 16 + l16][ks * 32 + hi * 16];
      union { uint4 q[2]; v16bf v; } un;
      un.q[0] = pp[0]; un.q[1] = pp[1];
      tb[ct][ks] = un.v;
    }
  }

  // ---- epilogue: out[b,t,v,c] = relu( Rt^T(v,kf) @ theta(kf,c) ) ----
  #pragma unroll
  for (int i = 0; i < 3; ++i) {
    int t = mh * 6 + m_set * 3 + i;
    #pragma unroll
    for (int r = 0; r < 8; ++r) {
      int f = r + hi * 8;
      #pragma unroll
      for (int k = 0; k < K_CHEB; ++k)
        RT[w][l16][k * 16 + f] = (__bf16)acc[i][k][r];
      RT[w][l16][48 + f] = (__bf16)0.f;
    }
    __syncthreads();
    v16bf ar[2];
    #pragma unroll
    for (int ks = 0; ks < 2; ++ks) {
      const uint4* pp = (const uint4*)&RT[w][l16][ks * 32 + hi * 16];
      union { uint4 q[2]; v16bf v; } un;
      un.q[0] = pp[0];
      un.q[1] = pp[1];
      ar[ks] = un.v;
    }
    #pragma unroll
    for (int ct = 0; ct < 4; ++ct) {
      v8f o = {};
      o = __builtin_amdgcn_wmma_f32_16x16x32_bf16(false, ar[0], false, tb[ct][0], (short)0, o, false, false);
      o = __builtin_amdgcn_wmma_f32_16x16x32_bf16(false, ar[1], false, tb[ct][1], (short)0, o, false, false);
      #pragma unroll
      for (int r = 0; r < 8; ++r) {
        int vg = vb + n_tile * 16 + r + hi * 8;
        float val = o[r] < 0.f ? 0.f : o[r];
        out[(((size_t)b * T_DIM + t) * V_DIM + vg) * C_DIM + ct * 16 + l16] = val;
      }
    }
    __syncthreads();
  }
}

// ---------------------------------------------------------------------------
extern "C" void kernel_launch(void* const* d_in, const int* in_sizes, int n_in,
                              void* d_out, int out_size, void* d_ws, size_t ws_size,
                              hipStream_t stream) {
  const float* x     = (const float*)d_in[0];
  const float* Att   = (const float*)d_in[1];
  const float* cheb  = (const float*)d_in[2];
  const float* theta = (const float*)d_in[3];
  float* out = (float*)d_out;
  __bf16* xt = (__bf16*)d_ws;   // 8*192*2048 bf16 ~= 6.3 MB of workspace

  const size_t nxt = (size_t)B_DIM * M_TOT * V_DIM;
  build_xt<<<dim3((unsigned)((nxt + 255) / 256)), dim3(256), 0, stream>>>(x, xt);
  cheb_fused<<<dim3(V_DIM / 32, 2, B_DIM), dim3(128), 0, stream>>>(xt, Att, cheb, theta, out);
}
